// scNET_19894288515587
// MI455X (gfx1250) — matrix operand order repaired and checked by reading, not compiled, not run on hardware
//
#include <hip/hip_runtime.h>
#include <math.h>

// ---------------------------------------------------------------------------
// scNET forward for MI455X (gfx1250): bf16 WMMA GEMMs + atomic graph scatters.
// ---------------------------------------------------------------------------

typedef __attribute__((ext_vector_type(16))) __bf16 v16bf;
typedef __attribute__((ext_vector_type(8)))  __bf16 v8bf;
typedef __attribute__((ext_vector_type(8)))  float  v8f;

#define LEAKY_SLOPE 0.01f
#define BM 128
#define BN 128
#define BK 32

__device__ __forceinline__ unsigned short f2bf(float f) {
  unsigned u = __float_as_uint(f);
  u += 0x7FFFu + ((u >> 16) & 1u);          // round-to-nearest-even
  return (unsigned short)(u >> 16);
}

__device__ __forceinline__ float actf(float v, int act) {
  if (act == 1) return v > 0.f ? v : LEAKY_SLOPE * v;   // leaky relu
  if (act == 2) return v > 0.f ? v : 0.f;               // relu
  return v;
}

// monotone float<->uint encoding for order-independent atomic max
__device__ __forceinline__ unsigned encf(float f) {
  unsigned u = __float_as_uint(f);
  return (u & 0x80000000u) ? ~u : (u | 0x80000000u);
}
__device__ __forceinline__ float decf(unsigned e) {
  unsigned u = (e & 0x80000000u) ? (e & 0x7FFFFFFFu) : ~e;
  return __uint_as_float(u);
}

// ---------------------------------------------------------------------------
// C[M,N] = act( A[M,K] @ W[K,N] + (accFlag ? C : 0) + (bias ? b : 0) )
// bf16 operands staged through LDS, f32 accumulation via v_wmma_f32_16x16x32_bf16
// block tile 128x128x32, 8 waves (2x4), each wave = 4x2 WMMA 16x16 tiles.
// Block-uniform fast path for interior tiles; global_prefetch of next K tile.
// ---------------------------------------------------------------------------
__global__ __launch_bounds__(256)
void gemm_bf16_wmma(const float* __restrict__ A, const float* __restrict__ W,
                    const float* __restrict__ bias, float* __restrict__ Cm,
                    int M, int N, int K, int accFlag, int act) {
  __shared__ unsigned short As[BM][40] __attribute__((aligned(32))); // row-major, pad 8
  __shared__ unsigned short Bs[BN][48] __attribute__((aligned(32))); // n-major,  pad 16

  const int tid  = threadIdx.x;
  const int lane = tid & 31;
  const int wave = tid >> 5;
  const int wm   = wave >> 2;   // 0..1  (M wave coord)
  const int wn   = wave & 3;    // 0..3  (N wave coord)
  const int l16  = lane & 15;
  const int g    = lane >> 4;   // lane half selects K sub-range

  const int m0 = blockIdx.y * BM;
  const int n0 = blockIdx.x * BN;
  const bool fullMN = (m0 + BM <= M) && (n0 + BN <= N);

  v8f acc[4][2];
  if (accFlag && fullMN) {
    #pragma unroll
    for (int mt = 0; mt < 4; ++mt)
      #pragma unroll
      for (int nt = 0; nt < 2; ++nt) {
        const int n = n0 + wn * 32 + nt * 16 + l16;
        #pragma unroll
        for (int r = 0; r < 8; ++r) {
          const int m = m0 + wm * 64 + mt * 16 + r + 8 * g;
          acc[mt][nt][r] = Cm[(size_t)m * N + n];
        }
      }
  } else {
    #pragma unroll
    for (int mt = 0; mt < 4; ++mt)
      #pragma unroll
      for (int nt = 0; nt < 2; ++nt) {
        const int n = n0 + wn * 32 + nt * 16 + l16;
        #pragma unroll
        for (int r = 0; r < 8; ++r) {
          const int m = m0 + wm * 64 + mt * 16 + r + 8 * g;
          float v = 0.f;
          if (accFlag && m < M && n < N) v = Cm[(size_t)m * N + n];
          acc[mt][nt][r] = v;
        }
      }
  }

  for (int k0 = 0; k0 < K; k0 += BK) {
    // ---- prefetch next K tile into caches (global_prefetch_b8) ----
    if (k0 + BK < K) {
      if (tid < BM) {                                   // A: 128 rows x 128B
        const int r = m0 + tid;
        if (r < M) __builtin_prefetch(A + (size_t)r * K + (k0 + BK), 0, 1);
      } else {                                          // W: 32 rows x 4 lines
        const int j = tid - BM, kr = j >> 2, sg = j & 3;
        const int kk = k0 + BK + kr;
        if (kk < K) __builtin_prefetch(W + (size_t)kk * N + n0 + sg * 32, 0, 1);
      }
    }
    __syncthreads();

    if (fullMN && (k0 + BK <= K)) {
      // ---- fast interior staging: no per-element guards ----
      #pragma unroll
      for (int t = 0; t < 4; ++t) {
        const int task = tid + t * 256;
        const int row  = task >> 3;
        const int c4   = (task & 7) << 2;
        const float4 f = *(const float4*)(A + (size_t)(m0 + row) * K + k0 + c4);
        ushort4 u; u.x = f2bf(f.x); u.y = f2bf(f.y); u.z = f2bf(f.z); u.w = f2bf(f.w);
        *(ushort4*)&As[row][c4] = u;
      }
      #pragma unroll
      for (int t = 0; t < 4; ++t) {
        const int task = tid + t * 256;
        const int kr   = task >> 5;
        const int c4   = (task & 31) << 2;
        const float4 f = *(const float4*)(W + (size_t)(k0 + kr) * N + n0 + c4);
        Bs[c4 + 0][kr] = f2bf(f.x);
        Bs[c4 + 1][kr] = f2bf(f.y);
        Bs[c4 + 2][kr] = f2bf(f.z);
        Bs[c4 + 3][kr] = f2bf(f.w);
      }
    } else {
      // ---- guarded staging for edge blocks / K tail ----
      #pragma unroll
      for (int t = 0; t < 4; ++t) {
        const int task = tid + t * 256;
        const int row  = task >> 3;
        const int c4   = (task & 7) << 2;
        const int gr = m0 + row, gc = k0 + c4;
        float v0 = 0.f, v1 = 0.f, v2 = 0.f, v3 = 0.f;
        if (gr < M) {
          const float* p = A + (size_t)gr * K + gc;
          if (gc + 3 < K) {
            float4 f = *(const float4*)p;
            v0 = f.x; v1 = f.y; v2 = f.z; v3 = f.w;
          } else {
            if (gc + 0 < K) v0 = p[0];
            if (gc + 1 < K) v1 = p[1];
            if (gc + 2 < K) v2 = p[2];
            if (gc + 3 < K) v3 = p[3];
          }
        }
        ushort4 u; u.x = f2bf(v0); u.y = f2bf(v1); u.z = f2bf(v2); u.w = f2bf(v3);
        *(ushort4*)&As[row][c4] = u;
      }
      #pragma unroll
      for (int t = 0; t < 4; ++t) {
        const int task = tid + t * 256;
        const int kr   = task >> 5;
        const int c4   = (task & 31) << 2;
        const int gk = k0 + kr, gn = n0 + c4;
        float v0 = 0.f, v1 = 0.f, v2 = 0.f, v3 = 0.f;
        if (gk < K) {
          const float* p = W + (size_t)gk * N + gn;
          if (gn + 3 < N) {
            float4 f = *(const float4*)p;
            v0 = f.x; v1 = f.y; v2 = f.z; v3 = f.w;
          } else {
            if (gn + 0 < N) v0 = p[0];
            if (gn + 1 < N) v1 = p[1];
            if (gn + 2 < N) v2 = p[2];
            if (gn + 3 < N) v3 = p[3];
          }
        }
        Bs[c4 + 0][kr] = f2bf(v0);
        Bs[c4 + 1][kr] = f2bf(v1);
        Bs[c4 + 2][kr] = f2bf(v2);
        Bs[c4 + 3][kr] = f2bf(v3);
      }
    }
    __syncthreads();

    // ---- A fragments: lane (l16,g) -> row l16, K = 16g .. 16g+15 contiguous ----
    v16bf af[4];
    #pragma unroll
    for (int mt = 0; mt < 4; ++mt) {
      const int row = wm * 64 + mt * 16 + l16;
      v8bf lo = *(const v8bf*)&As[row][g * 16];       // K = 16g .. 16g+7
      v8bf hi = *(const v8bf*)&As[row][g * 16 + 8];   // K = 16g+8 .. 16g+15
      af[mt] = __builtin_shufflevector(lo, hi, 0, 1, 2, 3, 4, 5, 6, 7,
                                       8, 9, 10, 11, 12, 13, 14, 15);
    }
    // ---- B fragments: lane n = l16, K = 16g .. 16g+15 contiguous ----
    v16bf bf_[2];
    #pragma unroll
    for (int nt = 0; nt < 2; ++nt) {
      const int n = wn * 32 + nt * 16 + l16;
      bf_[nt] = *(const v16bf*)&Bs[n][g * 16];
    }
    #pragma unroll
    for (int mt = 0; mt < 4; ++mt)
      #pragma unroll
      for (int nt = 0; nt < 2; ++nt)
        acc[mt][nt] = __builtin_amdgcn_wmma_f32_16x16x32_bf16(
            false, af[mt], false, bf_[nt], (short)0, acc[mt][nt], false, false);
  }

  // ---- epilogue: bias + activation + store ----
  if (fullMN) {
    #pragma unroll
    for (int mt = 0; mt < 4; ++mt)
      #pragma unroll
      for (int nt = 0; nt < 2; ++nt) {
        const int n = n0 + wn * 32 + nt * 16 + l16;
        const float bv = (bias != nullptr) ? bias[n] : 0.f;
        #pragma unroll
        for (int r = 0; r < 8; ++r) {
          const int m = m0 + wm * 64 + mt * 16 + r + 8 * g;
          Cm[(size_t)m * N + n] = actf(acc[mt][nt][r] + bv, act);
        }
      }
  } else {
    #pragma unroll
    for (int mt = 0; mt < 4; ++mt)
      #pragma unroll
      for (int nt = 0; nt < 2; ++nt) {
        const int n = n0 + wn * 32 + nt * 16 + l16;
        const float bv = (bias != nullptr && n < N) ? bias[n] : 0.f;
        #pragma unroll
        for (int r = 0; r < 8; ++r) {
          const int m = m0 + wm * 64 + mt * 16 + r + 8 * g;
          if (m < M && n < N)
            Cm[(size_t)m * N + n] = actf(acc[mt][nt][r] + bv, act);
        }
      }
  }
}

// ---------------------------------------------------------------------------
// Support kernels
// ---------------------------------------------------------------------------
__global__ __launch_bounds__(256)
void transpose_k(const float* __restrict__ in, float* __restrict__ out,
                 int R, int Cc) {
  __shared__ float tile[32][33];
  const int bx = blockIdx.x * 32, by = blockIdx.y * 32;
  const int tx = threadIdx.x & 31, ty = threadIdx.x >> 5;
  #pragma unroll
  for (int j = 0; j < 32; j += 8) {
    const int r = by + ty + j, c = bx + tx;
    tile[ty + j][tx] = (r < R && c < Cc) ? in[(size_t)r * Cc + c] : 0.f;
  }
  __syncthreads();
  #pragma unroll
  for (int j = 0; j < 32; j += 8) {
    const int r = bx + ty + j, c = by + tx;
    if (r < Cc && c < R) out[(size_t)r * R + c] = tile[tx][ty + j];
  }
}

__global__ void deg_kernel(const int* __restrict__ ei, int E, float* __restrict__ cnt) {
  const int e = blockIdx.x * blockDim.x + threadIdx.x;
  if (e < E) atomicAdd(&cnt[ei[E + e]], 1.0f);
}

__global__ void scatter_feat4(float* __restrict__ AGG, const float* __restrict__ X,
                              const int* __restrict__ ei, int E, int F) {
  const long long idx = (long long)blockIdx.x * blockDim.x + threadIdx.x;
  const int f4 = F >> 2;
  if (idx >= (long long)E * f4) return;
  const int e  = (int)(idx / f4);
  const int fo = (int)(idx % f4) << 2;
  const int s = ei[e], d = ei[E + e];
  const float* xs = X + (size_t)s * F + fo;
  float* ag = AGG + (size_t)d * F + fo;
  atomicAdd(&ag[0], xs[0]); atomicAdd(&ag[1], xs[1]);
  atomicAdd(&ag[2], xs[2]); atomicAdd(&ag[3], xs[3]);
}

__global__ void scale_rows(float* __restrict__ X, const float* __restrict__ cnt,
                           int Nn, int F) {
  const long long idx = (long long)blockIdx.x * blockDim.x + threadIdx.x;
  if (idx >= (long long)Nn * F) return;
  const int i = (int)(idx / F);
  X[idx] *= 1.0f / fmaxf(cnt[i], 1.0f);
}

__global__ void dinv_kernel(float* __restrict__ dinv, const float* __restrict__ cnt,
                            int Nn) {
  const int i = blockIdx.x * blockDim.x + threadIdx.x;
  if (i < Nn) dinv[i] = rsqrtf(fmaxf(cnt[i] + 1.0f, 1.0f));  // +1 self loop
}

__global__ void gcn_self(float* __restrict__ OUT, const float* __restrict__ HW,
                         const float* __restrict__ dinv, int Nn, int F) {
  const long long idx = (long long)blockIdx.x * blockDim.x + threadIdx.x;
  if (idx >= (long long)Nn * F) return;
  const int i = (int)(idx / F);
  OUT[idx] = HW[idx] * dinv[i] * dinv[i];
}

__global__ void gcn_scatter4(float* __restrict__ OUT, const float* __restrict__ HW,
                             const float* __restrict__ dinv, const int* __restrict__ ei,
                             int E, int F) {
  const long long idx = (long long)blockIdx.x * blockDim.x + threadIdx.x;
  const int f4 = F >> 2;
  if (idx >= (long long)E * f4) return;
  const int e  = (int)(idx / f4);
  const int fo = (int)(idx % f4) << 2;
  const int s = ei[e], d = ei[E + e];
  const float w = dinv[s] * dinv[d];
  const float* xs = HW + (size_t)s * F + fo;
  float* og = OUT + (size_t)d * F + fo;
  atomicAdd(&og[0], xs[0] * w); atomicAdd(&og[1], xs[1] * w);
  atomicAdd(&og[2], xs[2] * w); atomicAdd(&og[3], xs[3] * w);
}

__global__ void bias_act_k(float* __restrict__ X, const float* __restrict__ b,
                           int Nn, int F, int act) {
  const long long idx = (long long)blockIdx.x * blockDim.x + threadIdx.x;
  if (idx >= (long long)Nn * F) return;
  const int f = (int)(idx % F);
  X[idx] = actf(X[idx] + b[f], act);
}

// one wave32 per edge: alpha[e] = dot(Q[dst], K[src]) / sqrt(D)
__global__ __launch_bounds__(256)
void edge_dot(float* __restrict__ alpha, const float* __restrict__ Q,
              const float* __restrict__ Km, const int* __restrict__ ei,
              int E, int D) {
  const int wave = blockIdx.x * (blockDim.x >> 5) + (threadIdx.x >> 5);
  const int lane = threadIdx.x & 31;
  if (wave >= E) return;
  const int s = ei[wave], d = ei[E + wave];
  const float* q = Q + (size_t)d * D;
  const float* k = Km + (size_t)s * D;
  float acc = 0.f;
  for (int j = lane; j < D; j += 32) acc += q[j] * k[j];
  #pragma unroll
  for (int off = 16; off > 0; off >>= 1) acc += __shfl_xor(acc, off, 32);
  if (lane == 0) alpha[wave] = acc / sqrtf((float)D);
}

__global__ void init_enc(unsigned* __restrict__ m, int Nn) {
  const int i = blockIdx.x * blockDim.x + threadIdx.x;
  if (i < Nn) m[i] = 0x007FFFFFu;                      // enc(-inf)
}

__global__ void seg_max(unsigned* __restrict__ m, const float* __restrict__ alpha,
                        const int* __restrict__ ei, int E) {
  const int e = blockIdx.x * blockDim.x + threadIdx.x;
  if (e < E) atomicMax(&m[ei[E + e]], encf(alpha[e]));
}

__global__ void exp_sum(float* __restrict__ alpha, const unsigned* __restrict__ m,
                        float* __restrict__ z, const int* __restrict__ ei, int E) {
  const int e = blockIdx.x * blockDim.x + threadIdx.x;
  if (e >= E) return;
  const int d = ei[E + e];
  const float ex = expf(alpha[e] - decf(m[d]));
  alpha[e] = ex;
  atomicAdd(&z[d], ex);
}

__global__ void attn_soft4(float* __restrict__ OUT, const float* __restrict__ V,
                           const float* __restrict__ alpha, const float* __restrict__ z,
                           const int* __restrict__ ei, int E, int D) {
  const long long idx = (long long)blockIdx.x * blockDim.x + threadIdx.x;
  const int f4 = D >> 2;
  if (idx >= (long long)E * f4) return;
  const int e  = (int)(idx / f4);
  const int fo = (int)(idx % f4) << 2;
  const int s = ei[e], d = ei[E + e];
  const float w = alpha[e] / (z[d] + 1e-16f);
  const float* vs = V + (size_t)s * D + fo;
  float* og = OUT + (size_t)d * D + fo;
  atomicAdd(&og[0], vs[0] * w); atomicAdd(&og[1], vs[1] * w);
  atomicAdd(&og[2], vs[2] * w); atomicAdd(&og[3], vs[3] * w);
}

__global__ void attn_plain4(float* __restrict__ OUT, const float* __restrict__ V,
                            const float* __restrict__ alpha, const int* __restrict__ ei,
                            int E, int D) {
  const long long idx = (long long)blockIdx.x * blockDim.x + threadIdx.x;
  const int f4 = D >> 2;
  if (idx >= (long long)E * f4) return;
  const int e  = (int)(idx / f4);
  const int fo = (int)(idx % f4) << 2;
  const int s = ei[e], d = ei[E + e];
  const float w = alpha[e];
  const float* vs = V + (size_t)s * D + fo;
  float* og = OUT + (size_t)d * D + fo;
  atomicAdd(&og[0], vs[0] * w); atomicAdd(&og[1], vs[1] * w);
  atomicAdd(&og[2], vs[2] * w); atomicAdd(&og[3], vs[3] * w);
}

__global__ __launch_bounds__(256)
void red_sums(const float* __restrict__ a, int E, float* __restrict__ scal) {
  __shared__ float s1[256], s2[256];
  float x1 = 0.f, x2 = 0.f;
  for (int i = blockIdx.x * blockDim.x + threadIdx.x; i < E;
       i += gridDim.x * blockDim.x) {
    const float v = a[i];
    x1 += v; x2 += v * v;
  }
  s1[threadIdx.x] = x1; s2[threadIdx.x] = x2;
  __syncthreads();
  for (int st = 128; st > 0; st >>= 1) {
    if ((int)threadIdx.x < st) {
      s1[threadIdx.x] += s1[threadIdx.x + st];
      s2[threadIdx.x] += s2[threadIdx.x + st];
    }
    __syncthreads();
  }
  if (threadIdx.x == 0) {
    atomicAdd(&scal[0], s1[0]);
    atomicAdd(&scal[1], s2[0]);
  }
}

__global__ void sig_norm(float* __restrict__ a, int E, const float* __restrict__ scal,
                         float sp) {
  const int e = blockIdx.x * blockDim.x + threadIdx.x;
  if (e >= E) return;
  const float mean = scal[0] / (float)E;
  const float var  = (scal[1] - (float)E * mean * mean) / (float)(E - 1); // ddof=1
  const float t = (a[e] - mean) * sp * rsqrtf(fmaxf(var, 1e-30f));
  a[e] = 1.0f / (1.0f + expf(-t));
}

// ---------------------------------------------------------------------------
static inline unsigned nblk(long long t) { return (unsigned)((t + 255) / 256); }

extern "C" void kernel_launch(void* const* d_in, const int* in_sizes, int n_in,
                              void* d_out, int out_size, void* d_ws, size_t ws_size,
                              hipStream_t stream) {
  (void)in_sizes; (void)n_in; (void)out_size; (void)ws_size;
  constexpr int G = 3000, NC = 4096, EK = 65536, EP = 262144;
  constexpr int IR = 512, ER = 128, IC = 1024, EC = 256;

  const float* x     = (const float*)d_in[0];
  const int*   knn   = (const int*)d_in[1];
  const int*   ppi   = (const int*)d_in[2];
  const float* rl_W  = (const float*)d_in[3];
  const float* rl_b  = (const float*)d_in[4];
  const float* rs_Wl = (const float*)d_in[5];
  const float* rs_Wr = (const float*)d_in[6];
  const float* rs_b  = (const float*)d_in[7];
  const float* cs_Wl = (const float*)d_in[8];
  const float* cs_Wr = (const float*)d_in[9];
  const float* cs_b  = (const float*)d_in[10];
  const float* rg_W  = (const float*)d_in[11];
  const float* rg_b  = (const float*)d_in[12];
  const float* rq_W  = (const float*)d_in[13];
  const float* rq_b  = (const float*)d_in[14];
  const float* rk_W  = (const float*)d_in[15];
  const float* rk_b  = (const float*)d_in[16];
  const float* rv_W  = (const float*)d_in[17];
  const float* rv_b  = (const float*)d_in[18];
  const float* rsk_W = (const float*)d_in[19];
  const float* rsk_b = (const float*)d_in[20];
  const float* cg_W  = (const float*)d_in[21];
  const float* cg_b  = (const float*)d_in[22];
  const float* cq_W  = (const float*)d_in[23];
  const float* cq_b  = (const float*)d_in[24];
  const float* ck_W  = (const float*)d_in[25];
  const float* ck_b  = (const float*)d_in[26];
  const float* cv_W  = (const float*)d_in[27];
  const float* cv_b  = (const float*)d_in[28];
  const float* csk_W = (const float*)d_in[29];
  const float* csk_b = (const float*)d_in[30];
  const float* d1_W  = (const float*)d_in[31];
  const float* d1_b  = (const float*)d_in[32];
  const float* d2_W  = (const float*)d_in[33];
  const float* d2_b  = (const float*)d_in[34];
  const float* d3_W  = (const float*)d_in[35];
  const float* d3_b  = (const float*)d_in[36];

  float* out_row = (float*)d_out;                 // [G, ER]
  float* out_col = out_row + (size_t)G * ER;      // [NC, EC]
  float* out_rec = out_col + (size_t)NC * EC;     // [NC, G]

  // workspace layout (floats); big buffers are G*NC = 12,288,000 each
  float* H    = (float*)d_ws;
  float* T    = H + 12288000;
  float* AGG  = H + 24576000;
  float* S    = H + 36864000;
  float* CNT  = H + 49152000;      // 4096
  float* DINV = CNT + 4096;        // 4096
  float* SCAL = DINV + 4096;       // 2

  auto gemm = [&](float* Cm, const float* A, const float* W, const float* b,
                  int M, int N, int K, int acc, int act) {
    dim3 g((N + BN - 1) / BN, (M + BM - 1) / BM);
    gemm_bf16_wmma<<<g, 256, 0, stream>>>(A, W, b, Cm, M, N, K, acc, act);
  };

  // ---- encoder input linear: H = x @ rl_W + rl_b ----
  gemm(H, x, rl_W, rl_b, G, NC, NC, 0, 0);

  // ---- alternating col/row SAGE layers ----
  for (int i = 0; i < 2; ++i) {
    // columns (cells): T = H^T [NC,G]
    transpose_k<<<dim3((NC + 31) / 32, (G + 31) / 32), 256, 0, stream>>>(H, T, G, NC);
    hipMemsetAsync(CNT, 0, NC * sizeof(float), stream);
    deg_kernel<<<nblk(EK), 256, 0, stream>>>(knn, EK, CNT);
    hipMemsetAsync(AGG, 0, (size_t)NC * G * sizeof(float), stream);
    scatter_feat4<<<nblk((long long)EK * (G / 4)), 256, 0, stream>>>(AGG, T, knn, EK, G);
    scale_rows<<<nblk((long long)NC * G), 256, 0, stream>>>(AGG, CNT, NC, G);
    gemm(S, AGG, cs_Wl + (size_t)i * G * G, nullptr, NC, G, G, 0, 0);
    gemm(S, T, cs_Wr + (size_t)i * G * G, cs_b + (size_t)i * G, NC, G, G, 1, 1);
    transpose_k<<<dim3((G + 31) / 32, (NC + 31) / 32), 256, 0, stream>>>(S, H, NC, G);

    // rows (genes): H [G,NC]
    hipMemsetAsync(CNT, 0, G * sizeof(float), stream);
    deg_kernel<<<nblk(EP), 256, 0, stream>>>(ppi, EP, CNT);
    hipMemsetAsync(AGG, 0, (size_t)G * NC * sizeof(float), stream);
    scatter_feat4<<<nblk((long long)EP * (NC / 4)), 256, 0, stream>>>(AGG, H, ppi, EP, NC);
    scale_rows<<<nblk((long long)G * NC), 256, 0, stream>>>(AGG, CNT, G, NC);
    gemm(T, AGG, rs_Wl + (size_t)i * NC * NC, nullptr, G, NC, NC, 0, 0);
    gemm(T, H, rs_Wr + (size_t)i * NC * NC, rs_b + (size_t)i * NC, G, NC, NC, 1, 1);
    float* tmp = H; H = T; T = tmp;
  }

  // ---- row (gene) DimEncoder: GCN -> leaky -> TransformerConv(softmax) ----
  float* HW = AGG;                 // [G, IR]
  gemm(HW, H, rg_W, nullptr, G, IR, NC, 0, 0);
  hipMemsetAsync(CNT, 0, G * sizeof(float), stream);
  deg_kernel<<<nblk(EP), 256, 0, stream>>>(ppi, EP, CNT);
  dinv_kernel<<<nblk(G), 256, 0, stream>>>(DINV, CNT, G);
  float* R = S;                    // [G, IR]
  gcn_self<<<nblk((long long)G * IR), 256, 0, stream>>>(R, HW, DINV, G, IR);
  gcn_scatter4<<<nblk((long long)EP * (IR / 4)), 256, 0, stream>>>(R, HW, DINV, ppi, EP, IR);
  bias_act_k<<<nblk((long long)G * IR), 256, 0, stream>>>(R, rg_b, G, IR, 1);

  float* Qb    = AGG + 4194304;
  float* Kb    = AGG + 5242880;
  float* Vb    = AGG + 6291456;
  float* ALPHA = AGG + 7340032;
  gemm(Qb, R, rq_W, rq_b, G, ER, IR, 0, 0);
  gemm(Kb, R, rk_W, rk_b, G, ER, IR, 0, 0);
  gemm(Vb, R, rv_W, rv_b, G, ER, IR, 0, 0);
  gemm(out_row, R, rsk_W, rsk_b, G, ER, IR, 0, 0);          // root skip
  edge_dot<<<(EP + 7) / 8, 256, 0, stream>>>(ALPHA, Qb, Kb, ppi, EP, ER);
  init_enc<<<nblk(G), 256, 0, stream>>>((unsigned*)CNT, G);
  seg_max<<<nblk(EP), 256, 0, stream>>>((unsigned*)CNT, ALPHA, ppi, EP);
  hipMemsetAsync(DINV, 0, G * sizeof(float), stream);
  exp_sum<<<nblk(EP), 256, 0, stream>>>(ALPHA, (unsigned*)CNT, DINV, ppi, EP);
  attn_soft4<<<nblk((long long)EP * (ER / 4)), 256, 0, stream>>>(out_row, Vb, ALPHA,
                                                                 DINV, ppi, EP, ER);

  // ---- col (cell) DimEncoder: GCN -> leaky -> sigmoid-normalized attention ----
  transpose_k<<<dim3((NC + 31) / 32, (G + 31) / 32), 256, 0, stream>>>(H, T, G, NC);
  float* CW = AGG;                 // [NC, IC]
  gemm(CW, T, cg_W, nullptr, NC, IC, G, 0, 0);
  hipMemsetAsync(CNT, 0, NC * sizeof(float), stream);
  deg_kernel<<<nblk(EK), 256, 0, stream>>>(knn, EK, CNT);
  dinv_kernel<<<nblk(NC), 256, 0, stream>>>(DINV, CNT, NC);
  float* Cc = S;                   // [NC, IC]
  gcn_self<<<nblk((long long)NC * IC), 256, 0, stream>>>(Cc, CW, DINV, NC, IC);
  gcn_scatter4<<<nblk((long long)EK * (IC / 4)), 256, 0, stream>>>(Cc, CW, DINV, knn, EK, IC);
  bias_act_k<<<nblk((long long)NC * IC), 256, 0, stream>>>(Cc, cg_b, NC, IC, 1);

  gemm(Qb, Cc, cq_W, cq_b, NC, EC, IC, 0, 0);
  gemm(Kb, Cc, ck_W, ck_b, NC, EC, IC, 0, 0);
  gemm(Vb, Cc, cv_W, cv_b, NC, EC, IC, 0, 0);
  gemm(out_col, Cc, csk_W, csk_b, NC, EC, IC, 0, 0);        // root skip
  edge_dot<<<(EK + 7) / 8, 256, 0, stream>>>(ALPHA, Qb, Kb, knn, EK, EC);
  hipMemsetAsync(SCAL, 0, 2 * sizeof(float), stream);
  red_sums<<<256, 256, 0, stream>>>(ALPHA, EK, SCAL);
  sig_norm<<<nblk(EK), 256, 0, stream>>>(ALPHA, EK, SCAL, 3.0f);
  attn_plain4<<<nblk((long long)EK * (EC / 4)), 256, 0, stream>>>(out_col, Vb, ALPHA,
                                                                  knn, EK, EC);

  // ---- FeatureDecoder MLP ----
  float* D1 = AGG;                 // [NC, IC]
  float* D2 = AGG + 4194304;       // [NC, IC]
  gemm(D1, out_col, d1_W, d1_b, NC, IC, EC, 0, 2);
  gemm(D2, D1, d2_W, d2_b, NC, IC, IC, 0, 2);
  gemm(out_rec, D2, d3_W, d3_b, NC, G, IC, 0, 0);
}
